// VectorQuantize_41540923687460
// MI455X (gfx1250) — matrix operand order repaired and checked by reading, not compiled, not run on hardware
//
#include <hip/hip_runtime.h>
#include <hip/hip_bf16.h>

typedef __attribute__((ext_vector_type(16))) _Float16 v16h;
typedef __attribute__((ext_vector_type(8)))  _Float16 half8;
typedef __attribute__((ext_vector_type(8)))  float    v8f;

#define HCAT(lo,hi) __builtin_shufflevector((lo),(hi),0,1,2,3,4,5,6,7,8,9,10,11,12,13,14,15)

static __device__ inline v8f wmma16(v16h a, v16h b, v8f c) {
  return __builtin_amdgcn_wmma_f32_16x16x32_f16(false, a, false, b, (short)0, c, false, false);
}

static __device__ inline float redAdd32(float v) {
  #pragma unroll
  for (int m = 16; m > 0; m >>= 1) v += __shfl_xor(v, m, 32);
  return v;
}
static __device__ inline float redAdd16(float v) {
  #pragma unroll
  for (int m = 8; m > 0; m >>= 1) v += __shfl_xor(v, m, 16);
  return v;
}
static __device__ inline float redMax16(float v) {
  #pragma unroll
  for (int m = 8; m > 0; m >>= 1) v = fmaxf(v, __shfl_xor(v, m, 16));
  return v;
}

#define EPSF 1e-5f

// ---------------------------------------------------------------------------
// AdaIN + tokenize: target[b][c][n] (f32), x1/x2[b*256+n][c] = target + pos
// one wave per (b,c) pair; 8 pairs per 256-thread block
// ---------------------------------------------------------------------------
__global__ __launch_bounds__(256) void k_adain(
    const float* __restrict__ cF, const float* __restrict__ sF,
    const float* __restrict__ pos, float* __restrict__ target,
    float* __restrict__ x1, float* __restrict__ x2) {
  int tid = threadIdx.x, wid = tid >> 5, lane = tid & 31;
  int pair = blockIdx.x * 8 + wid;           // b*512 + ch
  int b = pair >> 9, ch = pair & 511;
  long base = (long)pair * 256;
  float sc = 0.f, sc2 = 0.f, ssu = 0.f, ss2 = 0.f;
  #pragma unroll
  for (int i = 0; i < 8; ++i) {
    int n = lane + i * 32;
    float cv = cF[base + n], sv = sF[base + n];
    sc += cv; sc2 += cv * cv; ssu += sv; ss2 += sv * sv;
  }
  sc = redAdd32(sc); sc2 = redAdd32(sc2); ssu = redAdd32(ssu); ss2 = redAdd32(ss2);
  float cm = sc * (1.f / 256.f), cvar = sc2 * (1.f / 256.f) - cm * cm;
  float sm = ssu * (1.f / 256.f), svar = ss2 * (1.f / 256.f) - sm * sm;
  float cs = sqrtf(cvar + EPSF), sstd = sqrtf(svar + EPSF);
  float rcs = sstd / cs;
  #pragma unroll
  for (int i = 0; i < 8; ++i) {
    int n = lane + i * 32;
    float tv = rcs * (cF[base + n] - cm) + sm;
    target[base + n] = tv;
    float tok = tv + pos[(long)n * 512 + ch];
    long xi = (long)(b * 256 + n) * 512 + ch;
    x1[xi] = tok; x2[xi] = tok;
  }
}

// ---------------------------------------------------------------------------
// generic transpose + f32->f16: src[L][K][N] -> dst[L][N][K]
// ---------------------------------------------------------------------------
__global__ __launch_bounds__(256) void k_tr(
    const float* __restrict__ src, _Float16* __restrict__ dst,
    int L, int K, int N) {
  long total = (long)L * K * N;
  for (long i = blockIdx.x * 256L + threadIdx.x; i < total; i += (long)gridDim.x * 256) {
    long kn = (long)K * N;
    int l = (int)(i / kn); long r = i - (long)l * kn;
    int k = (int)(r / N), n = (int)(r - (long)k * N);
    dst[((long)l * N + n) * K + k] = (_Float16)src[i];
  }
}

// enorm[j] = sum_c embed[c][j]^2 ; one wave per j
__global__ __launch_bounds__(256) void k_enorm(const float* __restrict__ embed,
                                               float* __restrict__ enorm) {
  int tid = threadIdx.x, wid = tid >> 5, lane = tid & 31;
  int j = blockIdx.x * 8 + wid;
  float s = 0.f;
  for (int c = lane; c < 512; c += 32) {
    float v = embed[(long)c * 1024 + j];
    s += v * v;
  }
  s = redAdd32(s);
  if (lane == 0) enorm[j] = s;
}

// ---------------------------------------------------------------------------
// shift_half + LayerNorm: x f32 [8192][512] -> h f16; one token row per block
// ---------------------------------------------------------------------------
__global__ __launch_bounds__(256) void k_shiftln(
    const float* __restrict__ x, _Float16* __restrict__ h,
    const float* __restrict__ gam, const float* __restrict__ bet) {
  __shared__ float red[256];
  int tid = threadIdx.x;
  long row = blockIdx.x;
  int n = (int)(row & 255);
  int c0 = tid, c1 = tid + 256;
  float v0 = (n > 0) ? x[(row - 1) * 512 + c0] : 0.f;   // c0 < 256: shifted half
  float v1 = x[row * 512 + c1];                         // c1 >= 256: unshifted
  red[tid] = v0 + v1; __syncthreads();
  #pragma unroll
  for (int s = 128; s > 0; s >>= 1) { if (tid < s) red[tid] += red[tid + s]; __syncthreads(); }
  float mean = red[0] * (1.f / 512.f); __syncthreads();
  red[tid] = v0 * v0 + v1 * v1; __syncthreads();
  #pragma unroll
  for (int s = 128; s > 0; s >>= 1) { if (tid < s) red[tid] += red[tid + s]; __syncthreads(); }
  float var = red[0] * (1.f / 512.f) - mean * mean;
  float inv = rsqrtf(var + EPSF);
  h[row * 512 + c0] = (_Float16)((v0 - mean) * inv * gam[c0] + bet[c0]);
  h[row * 512 + c1] = (_Float16)((v1 - mean) * inv * gam[c1] + bet[c1]);
}

// ---------------------------------------------------------------------------
// WMMA GEMM: C = A[M,K](f16) x Bt[N,K](f16)^T + bias
// mode 0: store f16   mode 1: io32 += (residual, f32)   mode 2: gelu -> f16
// block = 8 waves as 2x4 -> 64M x 256N ; wave tile 32M x 64N
// ---------------------------------------------------------------------------
__global__ __launch_bounds__(256) void k_gemm(
    const _Float16* __restrict__ A, const _Float16* __restrict__ Bt,
    const float* __restrict__ bias, _Float16* __restrict__ out16,
    float* __restrict__ io32, int M, int N, int K, int mode) {
  int tid = threadIdx.x, wid = tid >> 5, lane = tid & 31;
  int ln16 = lane & 15, hf = lane >> 4;
  int wy = wid >> 2, wx = wid & 3;
  int M0 = blockIdx.y * 64 + wy * 32;
  int N0 = blockIdx.x * 256 + wx * 64;
  v8f acc[2][4];
  #pragma unroll
  for (int i = 0; i < 2; ++i)
    #pragma unroll
    for (int j = 0; j < 4; ++j) acc[i][j] = (v8f){};
  for (int k0 = 0; k0 < K; k0 += 32) {
    v16h a[2];
    #pragma unroll
    for (int ms = 0; ms < 2; ++ms) {
      const _Float16* p = A + (long)(M0 + ms * 16 + ln16) * K + k0 + hf * 16;
      half8 lo = *(const half8*)p;
      half8 hi = *(const half8*)(p + 8);
      a[ms] = HCAT(lo, hi);
    }
    #pragma unroll
    for (int ns = 0; ns < 4; ++ns) {
      v16h b = *(const v16h*)(Bt + (long)(N0 + ns * 16 + ln16) * K + k0 + hf * 16);
      acc[0][ns] = wmma16(a[0], b, acc[0][ns]);
      acc[1][ns] = wmma16(a[1], b, acc[1][ns]);
    }
  }
  #pragma unroll
  for (int ms = 0; ms < 2; ++ms)
    #pragma unroll
    for (int ns = 0; ns < 4; ++ns) {
      int col = N0 + ns * 16 + ln16;
      float bi = bias[col];
      #pragma unroll
      for (int v = 0; v < 8; ++v) {
        int row = M0 + ms * 16 + v + hf * 8;
        float val = acc[ms][ns][v] + bi;
        long oi = (long)row * N + col;
        if (mode == 0) {
          out16[oi] = (_Float16)val;
        } else if (mode == 1) {
          io32[oi] += val;
        } else {
          float x3 = val * val * val;
          float g = 0.5f * val * (1.f + tanhf(0.7978845608028654f * (val + 0.044715f * x3)));
          out16[oi] = (_Float16)g;
        }
      }
    }
}

// ---------------------------------------------------------------------------
// Fused causal attention per (b, head): scores via WMMA (K padded 16->32),
// register softmax with 16-lane shuffle reductions, P@V via WMMA.
// block = 128 thr (4 waves), each wave owns 4 i-tiles. LDS = 40KB.
// ---------------------------------------------------------------------------
__global__ __launch_bounds__(128) void k_attn(const _Float16* __restrict__ qkv,
                                              _Float16* __restrict__ obuf) {
  __shared__ __align__(32) _Float16 vt[16 * 256];     // vt[d][tok]
  __shared__ __align__(32) _Float16 P[4][16 * 256];   // per-wave probs
  int blk = blockIdx.x;
  int b = blk >> 5, hd = blk & 31;
  int tid = threadIdx.x, wid = tid >> 5, lane = tid & 31;
  int ln16 = lane & 15, hf = lane >> 4;
  for (int i = tid; i < 4096; i += 128) {
    int d = i >> 8, tok = i & 255;
    vt[d * 256 + tok] = qkv[(long)(b * 256 + tok) * 1536 + 1024 + hd * 16 + d];
  }
  __syncthreads();
  half8 z8 = {};
  for (int itl = 0; itl < 4; ++itl) {
    int it = wid * 4 + itl;
    // q operand: rows it*16+r, dims 0..15 (pad K 16->32), pre-scaled by DH^-0.5
    const _Float16* qp = qkv + (long)(b * 256 + it * 16 + ln16) * 1536 + hd * 16 + hf * 8;
    half8 ql = *(const half8*)qp;
    ql = ql * (_Float16)0.25f;
    v16h aq = HCAT(ql, z8);
    v8f sc[16];
    #pragma unroll
    for (int jt = 0; jt < 16; ++jt) {
      // B: b[t] = K[8*hf + t][col] for t < 8; t >= 8 (k >= 16) is zero pad
      v16h bk = HCAT(*(const half8*)(qkv + (long)(b * 256 + jt * 16 + ln16) * 1536 + 512 + hd * 16 + hf * 8), z8);
      v8f zc = (v8f){};
      sc[jt] = wmma16(aq, bk, zc);
    }
    // causal mask + softmax (rows = it*16 + v + 8*hf, cols = jt*16 + ln16)
    #pragma unroll
    for (int v = 0; v < 8; ++v) {
      int row = it * 16 + v + hf * 8;
      float mx = -3.0e38f;
      #pragma unroll
      for (int jt = 0; jt < 16; ++jt) {
        int col = jt * 16 + ln16;
        float val = (col <= row) ? sc[jt][v] : -3.0e38f;
        sc[jt][v] = val;
        mx = fmaxf(mx, val);
      }
      mx = redMax16(mx);
      float sm = 0.f;
      #pragma unroll
      for (int jt = 0; jt < 16; ++jt) {
        float s = sc[jt][v];
        float e = (s > -1.0e30f) ? __expf(s - mx) : 0.f;
        sc[jt][v] = e;
        sm += e;
      }
      sm = redAdd16(sm);
      float inv = 1.f / sm;
      #pragma unroll
      for (int jt = 0; jt < 16; ++jt)
        P[wid][(v + hf * 8) * 256 + jt * 16 + ln16] = (_Float16)(sc[jt][v] * inv);
    }
    // o = P @ V : A = P[16][256], B = V[256][16] via vt (Bt layout)
    v8f oc = (v8f){};
    #pragma unroll
    for (int ks = 0; ks < 8; ++ks) {
      const _Float16* pp = &P[wid][ln16 * 256 + ks * 32 + hf * 16];
      half8 lo = *(const half8*)pp;
      half8 hi = *(const half8*)(pp + 8);
      v16h pa = HCAT(lo, hi);
      v16h vb = *(const v16h*)(&vt[ln16 * 256 + ks * 32 + hf * 16]);
      oc = wmma16(pa, vb, oc);
    }
    #pragma unroll
    for (int v = 0; v < 8; ++v) {
      long row = (long)(b * 256 + it * 16 + v + hf * 8);
      obuf[row * 512 + hd * 16 + ln16] = (_Float16)oc[v];
    }
  }
}

// t = 0.5*(x1+x2) -> f16
__global__ __launch_bounds__(256) void k_half(const float* __restrict__ x1,
                                              const float* __restrict__ x2,
                                              _Float16* __restrict__ th) {
  long i = blockIdx.x * 256L + threadIdx.x;
  th[i] = (_Float16)(0.5f * (x1[i] + x2[i]));
}

// ---------------------------------------------------------------------------
// VQ argmin: dist(p,j) = ||e_j||^2 - 2*t[p].e_j (+const). WMMA distance GEMM.
// block handles 16 rows x full 1024 codes; wave w covers 128 codes.
// ---------------------------------------------------------------------------
__global__ __launch_bounds__(256) void k_vq(
    const _Float16* __restrict__ th, const _Float16* __restrict__ embT,
    const float* __restrict__ enorm, int* __restrict__ ind,
    float* __restrict__ outInd) {
  __shared__ float sval[8][16];
  __shared__ int   sidx[8][16];
  int tid = threadIdx.x, wid = tid >> 5, lane = tid & 31;
  int ln16 = lane & 15, hf = lane >> 4;
  int mt = blockIdx.x;
  v8f acc[8];
  #pragma unroll
  for (int i = 0; i < 8; ++i) acc[i] = (v8f){};
  for (int k0 = 0; k0 < 512; k0 += 32) {
    const _Float16* ap = th + (long)(mt * 16 + ln16) * 512 + k0 + hf * 16;
    half8 lo = *(const half8*)ap;
    half8 hi = *(const half8*)(ap + 8);
    v16h a = HCAT(lo, hi);
    #pragma unroll
    for (int nt = 0; nt < 8; ++nt) {
      v16h b = *(const v16h*)(embT + (long)(wid * 128 + nt * 16 + ln16) * 512 + k0 + hf * 16);
      acc[nt] = wmma16(a, b, acc[nt]);
    }
  }
  float bv[8]; int bi[8];
  #pragma unroll
  for (int v = 0; v < 8; ++v) { bv[v] = 3.4e38f; bi[v] = 0x7fffffff; }
  #pragma unroll
  for (int nt = 0; nt < 8; ++nt) {
    int j = wid * 128 + nt * 16 + ln16;
    float e = enorm[j];
    #pragma unroll
    for (int v = 0; v < 8; ++v) {
      float val = e - 2.f * acc[nt][v];
      if (val < bv[v] || (val == bv[v] && j < bi[v])) { bv[v] = val; bi[v] = j; }
    }
  }
  #pragma unroll
  for (int v = 0; v < 8; ++v) {
    #pragma unroll
    for (int m = 8; m > 0; m >>= 1) {
      float ov = __shfl_xor(bv[v], m, 16);
      int   oi = __shfl_xor(bi[v], m, 16);
      if (ov < bv[v] || (ov == bv[v] && oi < bi[v])) { bv[v] = ov; bi[v] = oi; }
    }
  }
  if (ln16 == 0) {
    #pragma unroll
    for (int v = 0; v < 8; ++v) { sval[wid][v + hf * 8] = bv[v]; sidx[wid][v + hf * 8] = bi[v]; }
  }
  __syncthreads();
  if (tid < 16) {
    float best = sval[0][tid]; int bid = sidx[0][tid];
    #pragma unroll
    for (int w = 1; w < 8; ++w) {
      float ov = sval[w][tid]; int oi = sidx[w][tid];
      if (ov < best || (ov == best && oi < bid)) { best = ov; bid = oi; }
    }
    int row = mt * 16 + tid;
    ind[row] = bid;
    outInd[row] = (float)bid;
  }
}

// ---------------------------------------------------------------------------
// out = target + (quant - target); per-(b,c) partial sums for the losses.
// one wave per (b,c); deterministic (no atomics).
// ---------------------------------------------------------------------------
__global__ __launch_bounds__(256) void k_loss(
    const float* __restrict__ target, const float* __restrict__ embed,
    const int* __restrict__ ind, float* __restrict__ out,
    float* __restrict__ partials) {
  int tid = threadIdx.x, wid = tid >> 5, lane = tid & 31;
  int pair = blockIdx.x * 8 + wid;            // b*512 + ch
  int b = pair >> 9, ch = pair & 511;
  long base = (long)pair * 256;
  float sq = 0.f, sq2 = 0.f, st = 0.f, st2 = 0.f, sd = 0.f;
  #pragma unroll
  for (int i = 0; i < 8; ++i) {
    int n = lane + i * 32;
    int idx = ind[b * 256 + n];
    float q = embed[(long)ch * 1024 + idx];
    float t = target[base + n];
    out[base + n] = t + (q - t);
    float d = q - t;
    sq += q; sq2 += q * q; st += t; st2 += t * t; sd += d * d;
  }
  sq = redAdd32(sq); sq2 = redAdd32(sq2); st = redAdd32(st); st2 = redAdd32(st2); sd = redAdd32(sd);
  if (lane == 0) {
    float qm = sq * (1.f / 256.f), qv = sq2 * (1.f / 256.f) - qm * qm;
    float tm = st * (1.f / 256.f), tv = st2 * (1.f / 256.f) - tm * tm;
    float qs = sqrtf(qv + EPSF), ts = sqrtf(tv + EPSF);
    partials[(long)pair * 3 + 0] = sd;
    partials[(long)pair * 3 + 1] = (qm - tm) * (qm - tm);
    partials[(long)pair * 3 + 2] = (qs - ts) * (qs - ts);
  }
}

__global__ __launch_bounds__(256) void k_reduce(const float* __restrict__ partials,
                                                float* __restrict__ lossOut) {
  __shared__ float r0[256], r1[256], r2[256];
  int tid = threadIdx.x;
  float s0 = 0.f, s1 = 0.f, s2 = 0.f;
  for (int i = tid; i < 16384; i += 256) {
    s0 += partials[(long)i * 3 + 0];
    s1 += partials[(long)i * 3 + 1];
    s2 += partials[(long)i * 3 + 2];
  }
  r0[tid] = s0; r1[tid] = s1; r2[tid] = s2; __syncthreads();
  #pragma unroll
  for (int s = 128; s > 0; s >>= 1) {
    if (tid < s) { r0[tid] += r0[tid + s]; r1[tid] += r1[tid + s]; r2[tid] += r2[tid + s]; }
    __syncthreads();
  }
  if (tid == 0)
    lossOut[0] = r0[0] * (1.f / 4194304.f) + 10.f * (r1[0] + r2[0]) * (1.f / 16384.f);
}

// ---------------------------------------------------------------------------
extern "C" void kernel_launch(void* const* d_in, const int* in_sizes, int n_in,
                              void* d_out, int out_size, void* d_ws, size_t ws_size,
                              hipStream_t stream) {
  const float* cF    = (const float*)d_in[0];
  const float* sF    = (const float*)d_in[1];
  const float* embed = (const float*)d_in[2];
  const float* pos   = (const float*)d_in[3];
  const float* Wqkv  = (const float*)d_in[4];
  const float* bqkv  = (const float*)d_in[5];
  const float* Wo    = (const float*)d_in[6];
  const float* bo    = (const float*)d_in[7];
  const float* ln1g  = (const float*)d_in[8];
  const float* ln1b  = (const float*)d_in[9];
  const float* W1    = (const float*)d_in[10];
  const float* b1    = (const float*)d_in[11];
  const float* W2    = (const float*)d_in[12];
  const float* b2    = (const float*)d_in[13];
  const float* ln2g  = (const float*)d_in[14];
  const float* ln2b  = (const float*)d_in[15];

  size_t off = 0;
  auto alloc = [&](size_t bytes) -> void* {
    size_t o = (off + 255) & ~(size_t)255;
    off = o + bytes;
    return (void*)((char*)d_ws + o);
  };
  const long TOK = 8192;            // B*N
  float*    target = (float*)   alloc(4194304UL * 4);
  float*    x1     = (float*)   alloc(4194304UL * 4);
  float*    x2     = (float*)   alloc(4194304UL * 4);
  _Float16* hbuf   = (_Float16*)alloc(4194304UL * 2);
  _Float16* qkv    = (_Float16*)alloc(TOK * 1536UL * 2);
  _Float16* obuf   = (_Float16*)alloc(4194304UL * 2);
  _Float16* ffbuf  = (_Float16*)alloc(TOK * 2048UL * 2);
  _Float16* thbuf  = (_Float16*)alloc(4194304UL * 2);
  _Float16* WqkvT  = (_Float16*)alloc(8UL * 1536 * 512 * 2);
  _Float16* WoT    = (_Float16*)alloc(8UL * 512 * 512 * 2);
  _Float16* W1T    = (_Float16*)alloc(8UL * 2048 * 512 * 2);
  _Float16* W2T    = (_Float16*)alloc(8UL * 512 * 2048 * 2);
  _Float16* embT   = (_Float16*)alloc(1024UL * 512 * 2);
  float*    enorm  = (float*)   alloc(1024UL * 4);
  int*      ind    = (int*)     alloc(8192UL * 4);
  float*    parts  = (float*)   alloc(16384UL * 3 * 4);

  float* out     = (float*)d_out;              // [B,C,H,W] = 4194304
  float* outInd  = out + 4194304;              // 8192 indices (as float)
  float* outLoss = out + 4194304 + 8192;       // scalar

  // stage 0: AdaIN + tokens
  k_adain<<<2048, 256, 0, stream>>>(cF, sF, pos, target, x1, x2);

  // stage 1: weight transpose + f16 conversion (recomputed every call)
  k_tr<<<4096, 256, 0, stream>>>(Wqkv, WqkvT, 8, 512, 1536);
  k_tr<<<4096, 256, 0, stream>>>(Wo,   WoT,   8, 512, 512);
  k_tr<<<4096, 256, 0, stream>>>(W1,   W1T,   8, 512, 2048);
  k_tr<<<4096, 256, 0, stream>>>(W2,   W2T,   8, 2048, 512);
  k_tr<<<2048, 256, 0, stream>>>(embed, embT, 1, 512, 1024);
  k_enorm<<<128, 256, 0, stream>>>(embed, enorm);

  // stage 2: 8 reversible transformer layers
  for (int l = 0; l < 8; ++l) {
    k_shiftln<<<8192, 256, 0, stream>>>(x2, hbuf, ln1g + l * 512, ln1b + l * 512);
    k_gemm<<<dim3(6, 128), 256, 0, stream>>>(hbuf, WqkvT + (long)l * 1536 * 512,
                                             bqkv + l * 1536, qkv, nullptr,
                                             8192, 1536, 512, 0);
    k_attn<<<1024, 128, 0, stream>>>(qkv, obuf);
    k_gemm<<<dim3(2, 128), 256, 0, stream>>>(obuf, WoT + (long)l * 512 * 512,
                                             bo + l * 512, nullptr, x1,
                                             8192, 512, 512, 1);
    k_shiftln<<<8192, 256, 0, stream>>>(x1, hbuf, ln2g + l * 512, ln2b + l * 512);
    k_gemm<<<dim3(8, 128), 256, 0, stream>>>(hbuf, W1T + (long)l * 2048 * 512,
                                             b1 + l * 2048, ffbuf, nullptr,
                                             8192, 2048, 512, 2);
    k_gemm<<<dim3(2, 128), 256, 0, stream>>>(ffbuf, W2T + (long)l * 512 * 2048,
                                             b2 + l * 512, nullptr, x2,
                                             8192, 512, 2048, 1);
  }

  // stage 3: t = (x1+x2)/2 in f16, VQ argmin, outputs + loss
  k_half<<<16384, 256, 0, stream>>>(x1, x2, thbuf);
  k_vq<<<512, 256, 0, stream>>>(thbuf, embT, enorm, ind, outInd);
  k_loss<<<2048, 256, 0, stream>>>(target, embed, ind, out, parts);
  k_reduce<<<1, 256, 0, stream>>>(parts, outLoss);

  (void)in_sizes; (void)n_in; (void)out_size; (void)ws_size;
}
